// MutualInformationLoss_4612794876495
// MI455X (gfx1250) — matrix operand order repaired and checked by reading, not compiled, not run on hardware
//
#include <hip/hip_runtime.h>
#include <hip/hip_bf16.h>
#include <math.h>

typedef _Float16 v16h __attribute__((ext_vector_type(16)));
typedef _Float16 v8h  __attribute__((ext_vector_type(8)));
typedef float    v8f  __attribute__((ext_vector_type(8)));

#define BINS   23
#define PBINS  32
#define PRETERM 968.0f        // 1/(2*sigma^2), sigma = (1/22)*0.5
#define INV22  (1.0f/22.0f)
#define SMOOTH 1e-7f
#define WAVES_PER_BLOCK 8
#define BLOCKS_PER_BATCH 64

__global__ void mi_zero_kernel(float* __restrict__ acc, int n) {
    int i = blockIdx.x * blockDim.x + threadIdx.x;
    if (i < n) acc[i] = 0.0f;
}

__global__ __launch_bounds__(256)
void mi_pab_kernel(const float* __restrict__ pred,
                   const float* __restrict__ targ,
                   float* __restrict__ acc,   // [4][PBINS][PBINS]
                   int nvox) {
    // per-wave private transposed weight tiles: [wave][img][bin][voxel] f16
    __shared__ __align__(64) _Float16 ldsW[WAVES_PER_BLOCK][2][PBINS][32];
    __shared__ float lacc[PBINS * PBINS];

    const int tid   = threadIdx.x;
    const int wave  = tid >> 5;
    const int lane  = tid & 31;
    const int l     = lane & 15;
    const int half  = lane >> 4;
    const int batch = blockIdx.y;

    for (int i = tid; i < PBINS * PBINS; i += blockDim.x) lacc[i] = 0.0f;

    _Float16* wa = &ldsW[wave][0][0][0];
    _Float16* wb = &ldsW[wave][1][0][0];
    // pad rows (bins 23..31) stay zero for the whole kernel
    for (int j = BINS; j < PBINS; ++j) {
        wa[j * 32 + lane] = (_Float16)0.0f;
        wb[j * 32 + lane] = (_Float16)0.0f;
    }
    __syncthreads();

    const float* __restrict__ P = pred + (size_t)batch * (size_t)nvox;
    const float* __restrict__ T = targ + (size_t)batch * (size_t)nvox;

    v8f c00 = {}, c01 = {}, c10 = {}, c11 = {};

    const int chunks  = nvox >> 5;                    // 32 voxels per chunk
    const int wstride = gridDim.x * WAVES_PER_BLOCK;  // waves per batch

    for (int chunk = blockIdx.x * WAVES_PER_BLOCK + wave; chunk < chunks;
         chunk += wstride) {
        const int v = (chunk << 5) + lane;
        float x = P[v];
        float y = T[v];
        if (chunk + wstride < chunks) {   // uniform per wave
            __builtin_prefetch(&P[v + (wstride << 5)], 0, 1);
            __builtin_prefetch(&T[v + (wstride << 5)], 0, 1);
        }
        x = fminf(fmaxf(x, 0.0f), 1.0f);
        y = fminf(fmaxf(y, 0.0f), 1.0f);

        float ea[BINS], eb[BINS];
        float sa = 0.0f, sb = 0.0f;
#pragma unroll
        for (int j = 0; j < BINS; ++j) {
            float dx = x - (float)j * INV22;
            float dy = y - (float)j * INV22;
            float va = __expf(-PRETERM * dx * dx);
            float vb = __expf(-PRETERM * dy * dy);
            ea[j] = va; eb[j] = vb;
            sa += va;   sb += vb;
        }
        float ra = 1.0f / sa;
        float rb = 1.0f / sb;
#pragma unroll
        for (int j = 0; j < BINS; ++j) {
            wa[j * 32 + lane] = (_Float16)(ea[j] * ra);
            wb[j * 32 + lane] = (_Float16)(eb[j] * rb);
        }
        __builtin_amdgcn_wave_barrier();   // wave-synchronous LDS transpose

        // A fragments (16-bit A 16x32 layout): lane row = bin, two 8-half runs
        v8h a0lo = *reinterpret_cast<const v8h*>(&wa[l * 32 + half * 16]);
        v8h a0hi = *reinterpret_cast<const v8h*>(&wa[l * 32 + 8 + half * 16]);
        v8h a1lo = *reinterpret_cast<const v8h*>(&wa[(16 + l) * 32 + half * 16]);
        v8h a1hi = *reinterpret_cast<const v8h*>(&wa[(16 + l) * 32 + 8 + half * 16]);
        v16h A0 = __builtin_shufflevector(a0lo, a0hi, 0,1,2,3,4,5,6,7,8,9,10,11,12,13,14,15);
        v16h A1 = __builtin_shufflevector(a1lo, a1hi, 0,1,2,3,4,5,6,7,8,9,10,11,12,13,14,15);
        // B fragments (16-bit B 32x16 layout): lane col = bin, 16 contiguous K
        v16h B0 = *reinterpret_cast<const v16h*>(&wb[l * 32 + half * 16]);
        v16h B1 = *reinterpret_cast<const v16h*>(&wb[(16 + l) * 32 + half * 16]);

        c00 = __builtin_amdgcn_wmma_f32_16x16x32_f16(false, A0, false, B0, (short)0, c00, false, false);
        c01 = __builtin_amdgcn_wmma_f32_16x16x32_f16(false, A0, false, B1, (short)0, c01, false, false);
        c10 = __builtin_amdgcn_wmma_f32_16x16x32_f16(false, A1, false, B0, (short)0, c10, false, false);
        c11 = __builtin_amdgcn_wmma_f32_16x16x32_f16(false, A1, false, B1, (short)0, c11, false, false);
        __builtin_amdgcn_wave_barrier();
    }

    // fold per-wave C tiles into per-block LDS accumulator (ds_add_f32)
#pragma unroll
    for (int r = 0; r < 8; ++r) {
        int i0 = half * 8 + r;   // C layout: lanes 0-15 M=r, lanes 16-31 M=8+r
        atomicAdd(&lacc[i0 * PBINS + l],             c00[r]);
        atomicAdd(&lacc[i0 * PBINS + 16 + l],        c01[r]);
        atomicAdd(&lacc[(16 + i0) * PBINS + l],      c10[r]);
        atomicAdd(&lacc[(16 + i0) * PBINS + 16 + l], c11[r]);
    }
    __syncthreads();

    float* gacc = acc + (size_t)batch * PBINS * PBINS;
    for (int i = tid; i < PBINS * PBINS; i += blockDim.x)
        unsafeAtomicAdd(&gacc[i], lacc[i]);   // global_atomic_add_f32
}

__global__ __launch_bounds__(256)
void mi_final_kernel(const float* __restrict__ acc, float* __restrict__ out,
                     float invN) {
    __shared__ float pab[4][BINS][BINS];
    __shared__ float pa[4][BINS];
    __shared__ float pb[4][BINS];
    __shared__ float mi[4];
    const int tid = threadIdx.x;

    for (int idx = tid; idx < 4 * BINS * BINS; idx += blockDim.x) {
        int b = idx / (BINS * BINS);
        int r = idx % (BINS * BINS);
        int i = r / BINS, j = r % BINS;
        pab[b][i][j] = acc[b * PBINS * PBINS + i * PBINS + j] * invN;
    }
    if (tid < 4) mi[tid] = 0.0f;
    __syncthreads();

    // marginals = row/col sums of pab (weights sum to 1 over bins)
    for (int idx = tid; idx < 4 * BINS; idx += blockDim.x) {
        int b = idx / BINS, i = idx % BINS;
        float rs = 0.0f, cs = 0.0f;
        for (int j = 0; j < BINS; ++j) { rs += pab[b][i][j]; cs += pab[b][j][i]; }
        pa[b][i] = rs;
        pb[b][i] = cs;
    }
    __syncthreads();

    for (int idx = tid; idx < 4 * BINS * BINS; idx += blockDim.x) {
        int b = idx / (BINS * BINS);
        int r = idx % (BINS * BINS);
        int i = r / BINS, j = r % BINS;
        float p = pab[b][i][j];
        float q = pa[b][i] * pb[b][j];
        float t = p * logf((p + SMOOTH) / (q + SMOOTH) + SMOOTH);
        atomicAdd(&mi[b], t);
    }
    __syncthreads();
    if (tid == 0) out[0] = -0.25f * (mi[0] + mi[1] + mi[2] + mi[3]);
}

extern "C" void kernel_launch(void* const* d_in, const int* in_sizes, int n_in,
                              void* d_out, int out_size, void* d_ws, size_t ws_size,
                              hipStream_t stream) {
    const float* pred = (const float*)d_in[0];
    const float* targ = (const float*)d_in[1];
    float* out = (float*)d_out;
    float* acc = (float*)d_ws;           // 4 * 32 * 32 floats = 16 KB

    const int total = in_sizes[0];       // 4 * 96^3
    const int nvox  = total / 4;

    mi_zero_kernel<<<dim3((4 * PBINS * PBINS + 255) / 256), 256, 0, stream>>>(
        acc, 4 * PBINS * PBINS);
    mi_pab_kernel<<<dim3(BLOCKS_PER_BATCH, 4), 256, 0, stream>>>(
        pred, targ, acc, nvox);
    mi_final_kernel<<<1, 256, 0, stream>>>(acc, out, 1.0f / (float)nvox);
}
